// WindowedAssociativeLM_34368328303150
// MI455X (gfx1250) — matrix operand order repaired and checked by reading, not compiled, not run on hardware
//
#include <hip/hip_runtime.h>
#include <hip/hip_bf16.h>

// ---------------------------------------------------------------------------
// WindowedAssociativeLM on gfx1250 (MI455X).
//   1) convert weights/embedding to bf16 (GEMM operands; emb_bf16 L2-resident)
//   2) gather x = embedding[ids] as bf16
//   3) gx = x @ w_ih^T + b_ih              (WMMA bf16 -> f32)
//   4) GRU scan (f32, LDS h-state, K-split over 1024 threads)
//   5) h_emb = states @ wh^T + bh (WMMA, bf16 out); q/k projections (WMMA)
//   6) base = h_emb @ emb^T + out_bias -> d_out
//      (WMMA; 524MB output => store-bound; LDS-staged coalesced NT b128 stores)
//   7) banded softmax attention + gate, scatter-add into d_out
// ---------------------------------------------------------------------------

#define Bq 2
#define Lq 2048
#define Eq 256
#define Hq 512
#define Mq 128
#define Vq 32000
#define BL (Bq * Lq)          // 4096
#define G3 (3 * Hq)           // 1536

typedef __attribute__((ext_vector_type(16))) __bf16 v16bf;
typedef __attribute__((ext_vector_type(4)))  __bf16 v4bf;
typedef __attribute__((ext_vector_type(8)))  float  v8f;
typedef __attribute__((ext_vector_type(4)))  float  v4f;
typedef __attribute__((ext_vector_type(4)))  unsigned int u32x4;

union Frag { v16bf v; u32x4 q[2]; };

// ---------------------------------------------------------------------------
// f32 -> bf16 elementwise convert
// ---------------------------------------------------------------------------
__global__ void k_convert_bf16(const float* __restrict__ src,
                               __bf16* __restrict__ dst, int n) {
  int i = blockIdx.x * blockDim.x + threadIdx.x;
  if (i < n) dst[i] = (__bf16)src[i];
}

// w_hh [1536,512] -> w_hhT [512,1536] (coalesced GRU inner loop)
__global__ void k_transpose_whh(const float* __restrict__ whh,
                                float* __restrict__ whhT) {
  int i = blockIdx.x * blockDim.x + threadIdx.x;   // 786432 total
  if (i < G3 * Hq) {
    int g = i / Hq, kk = i % Hq;
    whhT[kk * G3 + g] = whh[i];
  }
}

// x_bf[pos, e] = bf16(embedding[ids[pos], e])
__global__ void k_gather_embed(const int* __restrict__ ids,
                               const float* __restrict__ emb,
                               __bf16* __restrict__ xb) {
  int pos = blockIdx.x;           // 4096
  int e   = threadIdx.x;          // 256
  int tok = ids[pos];
  xb[(size_t)pos * Eq + e] = (__bf16)emb[(size_t)tok * Eq + e];
}

// ---------------------------------------------------------------------------
// Generic WMMA GEMM:  C[M,N] = A[M,K] * B[N,K]^T + bias[N]
// A,B bf16 row-major, K % 32 == 0, M % 16 == 0, N % 128 == 0 (launch-exact).
// Block = 256 threads = 8 waves; each wave owns one 16x16 C tile; block covers
// 16(M) x 128(N). Fragment loads use contiguous k order, same for A and B:
//   A 16x32: lane(0-15) rows, K=0..15; lane(16-31) rows, K=16..31
//   B 32x16: lane(0-15) cols, K=0..15; lane(16-31) cols, K=16..31
// -> two contiguous 16B global_load_b128 per fragment per lane.
// Output tile is staged in LDS (row pitch 132 to avoid bank conflicts), then
// streamed out with fully-coalesced 16B stores: each wave writes one complete
// 512B output row per iteration.  NT=true adds the non-temporal hint so the
// 524MB logits stream does not evict the L2-resident embedding.
// ---------------------------------------------------------------------------
#define CPITCH 132

template <bool OUT_BF16, bool NT>
__global__ void k_wmma_gemm_nt(const __bf16* __restrict__ A,
                               const __bf16* __restrict__ Bm,
                               const float* __restrict__ bias,
                               void* __restrict__ Cout,
                               int M, int N, int K) {
  __shared__ float ctile[16 * CPITCH];   // 8.25 KB

  const int lane  = threadIdx.x & 31;
  const int wave  = threadIdx.x >> 5;
  const int lmod  = lane & 15;
  const int lhalf = lane >> 4;

  const int nblk = blockIdx.x * 128;
  const int n0   = nblk + wave * 16;
  const int m0   = blockIdx.y * 16;

  const __bf16* arow = A  + (size_t)(m0 + lmod) * K + lhalf * 16;
  const __bf16* brow = Bm + (size_t)(n0 + lmod) * K + lhalf * 16;

  v8f acc = {};
  for (int k0 = 0; k0 < K; k0 += 32) {
    Frag a, b;
    a.q[0] = *(const u32x4*)(arow + k0);        // K: base+0..7
    a.q[1] = *(const u32x4*)(arow + k0 + 8);    // K: base+8..15
    b.q[0] = *(const u32x4*)(brow + k0);        // K: base+0..7
    b.q[1] = *(const u32x4*)(brow + k0 + 8);    // K: base+8..15
    acc = __builtin_amdgcn_wmma_f32_16x16x32_bf16(
        /*neg_a=*/false, a.v, /*neg_b=*/false, b.v,
        /*c_mod=*/(short)0, acc, /*reuse_a=*/false, /*reuse_b=*/false);
  }

  // stage C tile (+bias) into LDS
  const float bv  = bias[n0 + lmod];
  const int ccol  = wave * 16 + lmod;
#pragma unroll
  for (int r = 0; r < 8; ++r)
    ctile[(r + lhalf * 8) * CPITCH + ccol] = acc[r] + bv;
  __syncthreads();

  // coalesced writeout: 512 float4 fragments; wave w, pass it -> row (w + it*8)
  const int t = threadIdx.x;
#pragma unroll
  for (int it = 0; it < 2; ++it) {
    const int f   = t + it * 256;    // 0..511
    const int row = f >> 5;          // 32 float4 per row
    const int c4  = (f & 31) * 4;
    const float* s = &ctile[row * CPITCH + c4];
    v4f val = { s[0], s[1], s[2], s[3] };
    if (OUT_BF16) {
      v4bf o = { (__bf16)val.x, (__bf16)val.y, (__bf16)val.z, (__bf16)val.w };
      v4bf* p = (v4bf*)((__bf16*)Cout + (size_t)(m0 + row) * N + nblk + c4);
      if (NT) __builtin_nontemporal_store(o, p); else *p = o;
    } else {
      v4f* p = (v4f*)((float*)Cout + (size_t)(m0 + row) * N + nblk + c4);
      if (NT) __builtin_nontemporal_store(val, p); else *p = val;
    }
  }
}

// ---------------------------------------------------------------------------
// GRU scan. One block per batch, 1024 threads: thread (p, t) with p = K-half,
// t = hidden index, computes partial dots of gate rows t, t+512, t+1024 over
// k in [p*256, p*256+256). Partials reduced through LDS; h lives in LDS.
// ---------------------------------------------------------------------------
__global__ void k_gru_scan(const float* __restrict__ gx,     // [B,L,3H]
                           const float* __restrict__ whhT,   // [H,3H]
                           const float* __restrict__ b_hh,   // [3H]
                           float* __restrict__ states,       // [B,L,H] f32
                           __bf16* __restrict__ states_bf) { // [B,L,H] bf16
  const int b   = blockIdx.x;
  const int tid = threadIdx.x;        // 0..1023
  const int t   = tid & (Hq - 1);
  const int p   = tid >> 9;           // wave-uniform K-partition

  __shared__ float h[Hq];
  __shared__ float parR[Hq], parZ[Hq], parN[Hq];
  if (p == 0) h[t] = 0.f;
  __syncthreads();

  const float bhr = b_hh[t], bhz = b_hh[Hq + t], bhn = b_hh[2 * Hq + t];
  const int kbeg = p * (Hq / 2);

  for (int l = 0; l < Lq; ++l) {
    float ar = 0.f, az = 0.f, an = 0.f;
#pragma unroll 8
    for (int kk = kbeg; kk < kbeg + Hq / 2; ++kk) {
      const float hk = h[kk];
      const float* w = whhT + (size_t)kk * G3;
      ar = fmaf(hk, w[t], ar);
      az = fmaf(hk, w[Hq + t], az);
      an = fmaf(hk, w[2 * Hq + t], an);
    }
    if (p) { parR[t] = ar; parZ[t] = az; parN[t] = an; }
    __syncthreads();                 // partials visible; all h reads done
    if (p == 0) {
      ar += parR[t]; az += parZ[t]; an += parN[t];
      const float* g = gx + ((size_t)(b * Lq + l)) * G3;
      const float xr = g[t], xz = g[Hq + t], xn = g[2 * Hq + t];
      const float r  = 1.f / (1.f + __expf(-(xr + ar + bhr)));
      const float z  = 1.f / (1.f + __expf(-(xz + az + bhz)));
      const float n  = tanhf(xn + r * (an + bhn));
      const float hn = (1.f - z) * n + z * h[t];
      h[t] = hn;
      const size_t o = ((size_t)(b * Lq + l)) * Hq + t;
      states[o]    = hn;
      states_bf[o] = (__bf16)hn;
    }
    __syncthreads();                 // h updated before next step's reads
  }
}

// ---------------------------------------------------------------------------
// Banded attention + gated scatter-add. One block per (b,i); 128 threads,
// thread j handles source position i-1-j. Each block owns output row (b,i),
// so atomics only resolve intra-block duplicate-token collisions.
// ---------------------------------------------------------------------------
__global__ void k_attn_scatter(const float* __restrict__ q,      // [BL,M]
                               const float* __restrict__ k,      // [BL,M]
                               const float* __restrict__ states, // [BL,H]
                               const float* __restrict__ wg,     // [H]
                               const float* __restrict__ bg,     // [1]
                               const float* __restrict__ mem_scale,
                               const int* __restrict__ ids,      // [BL]
                               float* __restrict__ out) {        // [BL,V]
  const int bi = blockIdx.x;       // 0..4095
  const int b  = bi >> 11;
  const int i  = bi & (Lq - 1);
  const int j  = threadIdx.x;      // 0..127

  __shared__ float qs[Mq];
  __shared__ float red[Mq];

  qs[j] = q[(size_t)bi * Mq + j];

  // gate = sigmoid(states[bi] . wg + bg)
  float gp = 0.f;
  for (int c = j; c < Hq; c += Mq) gp += states[(size_t)bi * Hq + c] * wg[c];
  red[j] = gp;
  __syncthreads();
  for (int s = 64; s > 0; s >>= 1) { if (j < s) red[j] += red[j + s]; __syncthreads(); }
  const float gate = 1.f / (1.f + __expf(-(red[0] + bg[0])));
  const float gmul = gate * mem_scale[0];
  __syncthreads();

  const int  jpos  = i - 1 - j;
  const bool valid = (jpos >= 0);
  float sc = -3.402823466e38f;
  if (valid) {
    const float* kr = k + ((size_t)(b * Lq + jpos)) * Mq;
    float d = 0.f;
#pragma unroll 8
    for (int m = 0; m < Mq; ++m) d = fmaf(qs[m], kr[m], d);
    sc = d * 0.088388347648318447f;    // 1/sqrt(128)
  }
  red[j] = sc;
  __syncthreads();
  for (int s = 64; s > 0; s >>= 1) { if (j < s) red[j] = fmaxf(red[j], red[j + s]); __syncthreads(); }
  const float mx = red[0];
  __syncthreads();
  const float e = valid ? __expf(sc - mx) : 0.f;
  red[j] = e;
  __syncthreads();
  for (int s = 64; s > 0; s >>= 1) { if (j < s) red[j] += red[j + s]; __syncthreads(); }
  const float sum = red[0];

  if (valid && sum > 0.f) {
    const float w = (e / sum) * gmul;
    const int tok = ids[b * Lq + jpos];
    atomicAdd(out + (size_t)bi * Vq + tok, w);
  }
}

// ---------------------------------------------------------------------------
extern "C" void kernel_launch(void* const* d_in, const int* in_sizes, int n_in,
                              void* d_out, int out_size, void* d_ws, size_t ws_size,
                              hipStream_t stream) {
  (void)in_sizes; (void)n_in; (void)out_size; (void)ws_size;

  const int*   ids      = (const int*)  d_in[0];
  const float* emb      = (const float*)d_in[1];
  const float* w_ih     = (const float*)d_in[2];
  const float* w_hh     = (const float*)d_in[3];
  const float* b_ih     = (const float*)d_in[4];
  const float* b_hh     = (const float*)d_in[5];
  const float* wq       = (const float*)d_in[6];
  const float* bq       = (const float*)d_in[7];
  const float* wk       = (const float*)d_in[8];
  const float* bk       = (const float*)d_in[9];
  const float* wg       = (const float*)d_in[10];
  const float* bg       = (const float*)d_in[11];
  const float* wh       = (const float*)d_in[12];
  const float* bh       = (const float*)d_in[13];
  const float* out_bias = (const float*)d_in[14];
  const float* mscale   = (const float*)d_in[15];
  float* out = (float*)d_out;

  char* ws = (char*)d_ws;
  size_t off = 0;
  auto alloc = [&](size_t bytes) -> char* {
    char* p = ws + off;
    off += (bytes + 255) & ~(size_t)255;
    return p;
  };
  __bf16* emb_bf   = (__bf16*)alloc((size_t)Vq * Eq * 2);   // 16.4 MB
  __bf16* wih_bf   = (__bf16*)alloc((size_t)G3 * Eq * 2);
  __bf16* wh_bf    = (__bf16*)alloc((size_t)Eq * Hq * 2);
  __bf16* wq_bf    = (__bf16*)alloc((size_t)Mq * Hq * 2);
  __bf16* wk_bf    = (__bf16*)alloc((size_t)Mq * Hq * 2);
  float*  whhT     = (float*) alloc((size_t)Hq * G3 * 4);   // 3 MB
  __bf16* x_bf     = (__bf16*)alloc((size_t)BL * Eq * 2);
  float*  gx       = (float*) alloc((size_t)BL * G3 * 4);   // 25 MB
  float*  states   = (float*) alloc((size_t)BL * Hq * 4);   // 8.4 MB
  __bf16* state_bf = (__bf16*)alloc((size_t)BL * Hq * 2);
  __bf16* hemb_bf  = (__bf16*)alloc((size_t)BL * Eq * 2);
  float*  qbuf     = (float*) alloc((size_t)BL * Mq * 4);
  float*  kbuf     = (float*) alloc((size_t)BL * Mq * 4);

  // 1) bf16 conversions (weights + embedding live in L2 thereafter)
  {
    int n;
    n = Vq * Eq; k_convert_bf16<<<(n + 255) / 256, 256, 0, stream>>>(emb, emb_bf, n);
    n = G3 * Eq; k_convert_bf16<<<(n + 255) / 256, 256, 0, stream>>>(w_ih, wih_bf, n);
    n = Eq * Hq; k_convert_bf16<<<(n + 255) / 256, 256, 0, stream>>>(wh, wh_bf, n);
    n = Mq * Hq; k_convert_bf16<<<(n + 255) / 256, 256, 0, stream>>>(wq, wq_bf, n);
    n = Mq * Hq; k_convert_bf16<<<(n + 255) / 256, 256, 0, stream>>>(wk, wk_bf, n);
  }
  k_transpose_whh<<<(G3 * Hq + 255) / 256, 256, 0, stream>>>(w_hh, whhT);

  // 2) embedding gather (bf16)
  k_gather_embed<<<BL, Eq, 0, stream>>>(ids, emb, x_bf);

  // 3) gx = x @ w_ih^T + b_ih   [4096 x 1536], K=256
  k_wmma_gemm_nt<false, false><<<dim3(G3 / 128, BL / 16), 256, 0, stream>>>(
      x_bf, wih_bf, b_ih, gx, BL, G3, Eq);

  // 4) GRU scan (1024 threads, K-split)
  k_gru_scan<<<Bq, 1024, 0, stream>>>(gx, whhT, b_hh, states, state_bf);

  // 5) projections
  //    h_emb (bf16 out)  [4096 x 256], K=512
  k_wmma_gemm_nt<true, false><<<dim3(Eq / 128, BL / 16), 256, 0, stream>>>(
      state_bf, wh_bf, bh, hemb_bf, BL, Eq, Hq);
  //    q, k              [4096 x 128], K=512
  k_wmma_gemm_nt<false, false><<<dim3(1, BL / 16), 256, 0, stream>>>(
      state_bf, wq_bf, bq, qbuf, BL, Mq, Hq);
  k_wmma_gemm_nt<false, false><<<dim3(1, BL / 16), 256, 0, stream>>>(
      state_bf, wk_bf, bk, kbuf, BL, Mq, Hq);

  // 6) base logits -> d_out  [4096 x 32000], K=256; NT coalesced b128 stores
  k_wmma_gemm_nt<false, true><<<dim3(Vq / 128, BL / 16), 256, 0, stream>>>(
      hemb_bf, emb_bf, out_bias, out, BL, Vq, Eq);

  // 7) banded attention + gated scatter-add into d_out
  k_attn_scatter<<<BL, Mq, 0, stream>>>(qbuf, kbuf, states, wg, bg, mscale, ids, out);
}
